// CustomAttention_7395933684170
// MI455X (gfx1250) — matrix operand order repaired and checked by reading, not compiled, not run on hardware
//
#include <hip/hip_runtime.h>

typedef __attribute__((ext_vector_type(16))) _Float16 v16h;
typedef __attribute__((ext_vector_type(8)))  _Float16 v8h;
typedef __attribute__((ext_vector_type(8)))  float    v8f;
typedef __attribute__((ext_vector_type(4)))  unsigned u32x4;
typedef __attribute__((ext_vector_type(8)))  int      i32x8;
typedef __attribute__((ext_vector_type(4)))  int      i32x4;

#define B_SZ   4
#define SEQ    2048
#define DIM    1024
#define HEADS  16
#define DHEAD  64
#define INNER  1024
#define QSCALE 0.125f   // 64^-0.5

__device__ __forceinline__ v16h cat16(v8h lo, v8h hi) {
  v16h r;
#pragma unroll
  for (int j = 0; j < 8; ++j) { r[j] = lo[j]; r[j + 8] = hi[j]; }
  return r;
}

__device__ __forceinline__ v8f wmma16(v16h a, v16h b, v8f c) {
  return __builtin_amdgcn_wmma_f32_16x16x32_f16(false, a, false, b, (short)0, c,
                                                false, false);
}

__device__ __forceinline__ v8f zero8() {
  v8f z = {0.f, 0.f, 0.f, 0.f, 0.f, 0.f, 0.f, 0.f};
  return z;
}

__device__ __forceinline__ unsigned pack2(_Float16 a, _Float16 b) {
  union { _Float16 h[2]; unsigned u; } p;
  p.h[0] = a; p.h[1] = b;
  return p.u;
}

// TDM 2D tile load: global[tile_y rows x tile_x elems, 2B] -> LDS, with
// optional LDS row padding. D# built per CDNA5 ISA 8.3/8.4 (group0 + group1).
__device__ __forceinline__ void tdm_load_2d_f16(
    unsigned lds_byte_addr, const void* gaddr,
    unsigned tile_x, unsigned tile_y,
    unsigned tensor_x, unsigned tensor_y, unsigned stride_x,
    unsigned pad_interval_code, unsigned pad_amount_code, bool pad_en) {
  unsigned long long ga = (unsigned long long)gaddr;
  u32x4 g0;
  g0[0] = 1u;                                    // count=1 (valid), user mode
  g0[1] = lds_byte_addr;                         // lds_addr
  g0[2] = (unsigned)(ga & 0xFFFFFFFFu);          // global_addr[31:0]
  g0[3] = (unsigned)((ga >> 32) & 0x01FFFFFFu)   // global_addr[56:32]
          | (2u << 30);                          // type=2 ("image")
  unsigned d0 = (1u << 16);                      // data_size=1 -> 2 bytes
  if (pad_en)
    d0 |= (1u << 20) | (pad_interval_code << 22) | (pad_amount_code << 25);
  i32x8 g1;
  g1[0] = (int)d0;
  g1[1] = (int)((tensor_x & 0xFFFFu) << 16);                 // dim0[15:0]
  g1[2] = (int)(((tensor_x >> 16) & 0xFFFFu) |
                ((tensor_y & 0xFFFFu) << 16));               // dim0 hi | dim1 lo
  g1[3] = (int)(((tensor_y >> 16) & 0xFFFFu) |
                ((tile_x & 0xFFFFu) << 16));                 // dim1 hi | tile0
  g1[4] = (int)(tile_y & 0xFFFFu);                           // tile1, tile2=0
  g1[5] = (int)stride_x;                                     // dim0_stride lo
  g1[6] = 0;                                                 // stride hi
  g1[7] = 0;
  i32x4 z4 = {0, 0, 0, 0};
  i32x8 z8 = {0, 0, 0, 0, 0, 0, 0, 0};
  __builtin_amdgcn_tensor_load_to_lds(g0, g1, z4, z4, z8, 0);
}

// ---------------------------------------------------------------- cast kernel
__global__ void cast_f32_to_f16(const float* __restrict__ s,
                                _Float16* __restrict__ d, int n) {
  int i = (blockIdx.x * blockDim.x + threadIdx.x) * 8;
  if (i >= n) return;
  float4 a = *(const float4*)(s + i);
  float4 b = *(const float4*)(s + i + 4);
  v8h r;
  r[0] = (_Float16)a.x; r[1] = (_Float16)a.y;
  r[2] = (_Float16)a.z; r[3] = (_Float16)a.w;
  r[4] = (_Float16)b.x; r[5] = (_Float16)b.y;
  r[6] = (_Float16)b.z; r[7] = (_Float16)b.w;
  *(v8h*)(d + i) = r;
}

// ----------------------------------------------------------------- WMMA GEMM
template <bool PROJ>
__global__ __launch_bounds__(256) void gemm_wmma_kernel(
    const _Float16* __restrict__ A, const _Float16* __restrict__ Bm,
    int M, int Nw, int K,
    _Float16* __restrict__ qh, _Float16* __restrict__ kh,
    _Float16* __restrict__ vh,
    const float* __restrict__ bias, float* __restrict__ Cout) {
  __shared__ __align__(16) _Float16 As[128][40];  // [m][k], +8 pad
  __shared__ __align__(16) _Float16 Bs[128][40];  // [n][k] (transposed), +8 pad

  const int tid   = threadIdx.x;
  const int lane  = tid & 31;
  const int l15   = lane & 15;
  const int half  = lane >> 4;
  const int wave  = tid >> 5;
  const int waveM = wave >> 2;
  const int waveN = wave & 3;
  const int bm0 = blockIdx.y * 128;
  const int bn0 = blockIdx.x * 128;

  v8f acc[4][2];
#pragma unroll
  for (int i = 0; i < 4; ++i)
#pragma unroll
    for (int j = 0; j < 2; ++j) acc[i][j] = zero8();

  for (int k0 = 0; k0 < K; k0 += 32) {
#pragma unroll
    for (int t = 0; t < 2; ++t) {
      int idx = tid + t * 256;
      int r = idx >> 2;
      int c = (idx & 3) << 3;
      const _Float16* gp = A + (size_t)(bm0 + r) * K + k0 + c;
      *(float4*)&As[r][c] = *(const float4*)gp;
      if (k0 + 32 < K) __builtin_prefetch(gp + 32, 0, 1);
    }
    {
      int kp = tid >> 4;
      int nn = (tid & 15) << 3;
      const _Float16* g0 = Bm + (size_t)(k0 + 2 * kp) * Nw + bn0 + nn;
      const _Float16* g1 = g0 + Nw;
      float4 e = *(const float4*)g0;
      float4 o = *(const float4*)g1;
      if (k0 + 32 < K) __builtin_prefetch(g0 + (size_t)32 * Nw, 0, 1);
      const _Float16* eh = (const _Float16*)&e;
      const _Float16* oh = (const _Float16*)&o;
#pragma unroll
      for (int j = 0; j < 8; ++j)
        *(unsigned*)&Bs[nn + j][2 * kp] = pack2(eh[j], oh[j]);
    }
    __syncthreads();

    v16h af[4];
#pragma unroll
    for (int i = 0; i < 4; ++i) {
      int row = waveM * 64 + i * 16 + l15;
      v8h lo = *(const v8h*)&As[row][half * 16];
      v8h hi = *(const v8h*)&As[row][half * 16 + 8];
      af[i] = cat16(lo, hi);
    }
    v16h bf[2];
#pragma unroll
    for (int j = 0; j < 2; ++j) {
      int col = waveN * 32 + j * 16 + l15;
      v8h lo = *(const v8h*)&Bs[col][half * 16];
      v8h hi = *(const v8h*)&Bs[col][half * 16 + 8];
      bf[j] = cat16(lo, hi);
    }
#pragma unroll
    for (int i = 0; i < 4; ++i)
#pragma unroll
      for (int j = 0; j < 2; ++j) acc[i][j] = wmma16(af[i], bf[j], acc[i][j]);
    __syncthreads();
  }

#pragma unroll
  for (int i = 0; i < 4; ++i) {
#pragma unroll
    for (int j = 0; j < 2; ++j) {
      int gcol = bn0 + waveN * 32 + j * 16 + l15;
      if (PROJ) {
        float bv = bias[gcol];
#pragma unroll
        for (int vg = 0; vg < 8; ++vg) {
          int t = bm0 + waveM * 64 + i * 16 + half * 8 + vg;
          Cout[(size_t)t * Nw + gcol] = acc[i][j][vg] + bv;
        }
      } else {
        int which  = gcol >> 10;
        int within = gcol & 1023;
        int h = within >> 6;
        int d = within & 63;
        _Float16* dst = (which == 0) ? qh : (which == 1) ? kh : vh;
        float sc = (which == 0) ? QSCALE : 1.0f;
#pragma unroll
        for (int vg = 0; vg < 8; ++vg) {
          int t  = bm0 + waveM * 64 + i * 16 + half * 8 + vg;
          int bb = t >> 11;
          int nn = t & 2047;
          dst[(((size_t)bb * HEADS + h) * SEQ + nn) * DHEAD + d] =
              (_Float16)(acc[i][j][vg] * sc);
        }
      }
    }
  }
}

// ----------------------------------------------------------- flash attention
// S^T = K*Q^T, O^T = V^T*P^T (per-lane softmax stats, one shfl_xor(16)).
// K tiles are DMA'd by the Tensor Data Mover into a double-buffered LDS tile
// (padded rows via D# pad fields); tile i+1 streams while tile i computes.
__global__ __launch_bounds__(128) void attn_wmma_kernel(
    const _Float16* __restrict__ qh, const _Float16* __restrict__ kh,
    const _Float16* __restrict__ vh, _Float16* __restrict__ oh) {
  __shared__ __align__(16) _Float16 Ks2[2][64][72];  // [buf][key][d], TDM dest
  __shared__ __align__(16) _Float16 Vt[64][72];      // [d][key]
  __shared__ __align__(16) _Float16 Ps[4][16][72];   // per-wave P^T [query][key]

  const int tid  = threadIdx.x;
  const int lane = tid & 31;
  const int l15  = lane & 15;
  const int half = lane >> 4;
  const int wave = tid >> 5;

  const int bh = blockIdx.y;
  const int q0 = blockIdx.x * 64;
  const size_t base = (size_t)bh * SEQ * DHEAD;
  const int NB = SEQ / 64;

  // Q as B-fragments: column = query l15; k = d (2 chunks of 32)
  v16h qb[2];
  {
    const _Float16* qp = qh + base + (size_t)(q0 + wave * 16 + l15) * DHEAD;
#pragma unroll
    for (int c = 0; c < 2; ++c) {
      v8h lo = *(const v8h*)(qp + c * 32 + half * 16);
      v8h hi = *(const v8h*)(qp + c * 32 + half * 16 + 8);
      qb[c] = cat16(lo, hi);
    }
  }

  // TDM prologue: stream K tile 0 into buffer 0.
  // pad: 4 dwords (8 halves) after every 32 dwords (64 halves) -> 72-half rows
  if (tid < 32) {
    tdm_load_2d_f16((unsigned)(unsigned long long)&Ks2[0][0][0],
                    kh + base, /*tile_x=*/DHEAD, /*tile_y=*/64,
                    /*tensor_x=*/DHEAD, /*tensor_y=*/SEQ, /*stride_x=*/DHEAD,
                    /*pad_interval=*/4u, /*pad_amount=*/3u, true);
  }

  v8f o[4];
#pragma unroll
  for (int dt = 0; dt < 4; ++dt) o[dt] = zero8();
  float rmax = -3.0e38f, rsum = 0.f;

  for (int it = 0; it < NB; ++it) {
    const int kb = it * 64;
    const int buf = it & 1;

    // V tile transposed to [d][key] (manual; TDM cannot transpose)
#pragma unroll
    for (int t = 0; t < 2; ++t) {
      int idx = tid + t * 128;
      int rp = idx >> 3;
      int cc = (idx & 7) << 3;
      const _Float16* g0 = vh + base + (size_t)(kb + 2 * rp) * DHEAD + cc;
      float4 e = *(const float4*)g0;
      float4 od = *(const float4*)(g0 + DHEAD);
      const _Float16* eh = (const _Float16*)&e;
      const _Float16* oh2 = (const _Float16*)&od;
#pragma unroll
      for (int j = 0; j < 8; ++j)
        *(unsigned*)&Vt[cc + j][2 * rp] = pack2(eh[j], oh2[j]);
    }

    // kick off next K tile, then wait for the current one
    if (tid < 32) {
      if (it + 1 < NB) {
        tdm_load_2d_f16((unsigned)(unsigned long long)&Ks2[buf ^ 1][0][0],
                        kh + base + (size_t)(kb + 64) * DHEAD, DHEAD, 64,
                        DHEAD, SEQ, DHEAD, 4u, 3u, true);
        __builtin_amdgcn_s_wait_tensorcnt(1);
      } else {
        __builtin_amdgcn_s_wait_tensorcnt(0);
      }
    }
    __syncthreads();

    // S^T = K * Q^T
    v8f s[4];
#pragma unroll
    for (int nt = 0; nt < 4; ++nt) {
      v8f z = zero8();
      int krow = nt * 16 + l15;
#pragma unroll
      for (int c = 0; c < 2; ++c) {
        v8h lo = *(const v8h*)&Ks2[buf][krow][c * 32 + half * 16];
        v8h hi = *(const v8h*)&Ks2[buf][krow][c * 32 + half * 16 + 8];
        z = wmma16(cat16(lo, hi), qb[c], z);
      }
      s[nt] = z;
    }

    // online softmax: per-lane over 32 key values + one half-swap shuffle
    float m = -3.0e38f;
#pragma unroll
    for (int nt = 0; nt < 4; ++nt)
#pragma unroll
      for (int i = 0; i < 8; ++i) m = fmaxf(m, s[nt][i]);
    m = fmaxf(m, __shfl_xor(m, 16, 32));
    float nm = fmaxf(rmax, m);
    float corr = __expf(rmax - nm);
    rmax = nm;
    float lsum = 0.f;
#pragma unroll
    for (int nt = 0; nt < 4; ++nt)
#pragma unroll
      for (int i = 0; i < 8; ++i) {
        float p = __expf(s[nt][i] - nm);
        s[nt][i] = p;
        lsum += p;
      }
    lsum += __shfl_xor(lsum, 16, 32);
    rsum = rsum * corr + lsum;
#pragma unroll
    for (int dt = 0; dt < 4; ++dt)
#pragma unroll
      for (int vg = 0; vg < 8; ++vg) o[dt][vg] *= corr;

    // stage P^T rows: 8 consecutive keys -> one b128 store per tile
#pragma unroll
    for (int nt = 0; nt < 4; ++nt) {
      v8h pk;
#pragma unroll
      for (int i = 0; i < 8; ++i) pk[i] = (_Float16)s[nt][i];
      *(v8h*)&Ps[wave][l15][nt * 16 + half * 8] = pk;
    }
    __syncthreads();

    // O^T += V^T * P^T
#pragma unroll
    for (int c = 0; c < 2; ++c) {
      v8h plo = *(const v8h*)&Ps[wave][l15][c * 32 + half * 16];
      v8h phi = *(const v8h*)&Ps[wave][l15][c * 32 + half * 16 + 8];
      v16h pb = cat16(plo, phi);
#pragma unroll
      for (int dt = 0; dt < 4; ++dt) {
        int drow = dt * 16 + l15;
        v8h alo = *(const v8h*)&Vt[drow][c * 32 + half * 16];
        v8h ahi = *(const v8h*)&Vt[drow][c * 32 + half * 16 + 8];
        o[dt] = wmma16(cat16(alo, ahi), pb, o[dt]);
      }
    }
    __syncthreads();
  }

  const int hh = bh & (HEADS - 1);
  const int bb = bh >> 4;
  float inv = 1.0f / rsum;
  int token = q0 + wave * 16 + l15;
  size_t rb = ((size_t)bb * SEQ + token) * INNER + (size_t)hh * DHEAD;
#pragma unroll
  for (int dt = 0; dt < 4; ++dt) {
    v8h r;
#pragma unroll
    for (int vg = 0; vg < 8; ++vg) r[vg] = (_Float16)(o[dt][vg] * inv);
    *(v8h*)(oh + rb + dt * 16 + half * 8) = r;
  }
}

// ------------------------------------------------------------------ launcher
extern "C" void kernel_launch(void* const* d_in, const int* in_sizes, int n_in,
                              void* d_out, int out_size, void* d_ws,
                              size_t ws_size, hipStream_t stream) {
  const float* x     = (const float*)d_in[0];
  const float* w_qkv = (const float*)d_in[1];
  const float* w_out = (const float*)d_in[2];
  const float* b_out = (const float*)d_in[3];
  float* out = (float*)d_out;

  const size_t NX  = (size_t)B_SZ * SEQ * DIM;
  const size_t NWQ = (size_t)DIM * 3 * INNER;
  const size_t NWO = (size_t)INNER * DIM;
  const size_t NQ  = (size_t)B_SZ * HEADS * SEQ * DHEAD;

  _Float16* xh  = (_Float16*)d_ws;
  _Float16* wqh = xh + NX;
  _Float16* woh = wqh + NWQ;
  _Float16* q_h = woh + NWO;
  _Float16* k_h = q_h + NQ;
  _Float16* v_h = k_h + NQ;
  _Float16* a_h = v_h + NQ;

  cast_f32_to_f16<<<dim3((unsigned)(NX  / 2048)), 256, 0, stream>>>(x, xh, (int)NX);
  cast_f32_to_f16<<<dim3((unsigned)(NWQ / 2048)), 256, 0, stream>>>(w_qkv, wqh, (int)NWQ);
  cast_f32_to_f16<<<dim3((unsigned)(NWO / 2048)), 256, 0, stream>>>(w_out, woh, (int)NWO);

  gemm_wmma_kernel<false><<<dim3((3 * INNER) / 128, (B_SZ * SEQ) / 128), 256, 0,
                            stream>>>(xh, wqh, B_SZ * SEQ, 3 * INNER, DIM, q_h,
                                      k_h, v_h, nullptr, nullptr);

  attn_wmma_kernel<<<dim3(SEQ / 64, B_SZ * HEADS), 128, 0, stream>>>(q_h, k_h,
                                                                     v_h, a_h);

  gemm_wmma_kernel<true><<<dim3(DIM / 128, (B_SZ * SEQ) / 128), 256, 0,
                           stream>>>(a_h, woh, B_SZ * SEQ, DIM, INNER, nullptr,
                                     nullptr, nullptr, b_out, out);
}
